// cls_model_57767310131835
// MI455X (gfx1250) — matrix operand order repaired and checked by reading, not compiled, not run on hardware
//
#include <hip/hip_runtime.h>
#include <hip/hip_bf16.h>

// ---------------- problem constants ----------------
constexpr int B_    = 16;
constexpr int N_    = 2048;
constexpr int FIN_  = 6;
constexpr int FOUT_ = 128;
constexpr int KCH_  = 6;
constexpr int CLS_  = 40;
constexpr int KF_   = KCH_ * FIN_;   // 36
constexpr int NT_   = N_ / 16;       // 128 row tiles

typedef __attribute__((ext_vector_type(2)))  float  v2f;
typedef __attribute__((ext_vector_type(8)))  float  v8f;
typedef __attribute__((ext_vector_type(8)))  __bf16 v8bf;
typedef __attribute__((ext_vector_type(16))) __bf16 v16bf;

#define DEVI __device__ __forceinline__

// ---------------------------------------------------------------------------
// Inner-product 16x16 tile via two V_WMMA_F32_16X16X4_F32 (K = coords 0..7,
// coords 6,7 zero-padded). Returns inner[i0+M][j0+n] in C/D layout:
// acc[r] = inner[i0 + r + 8*(lane/16)][j0 + lane%16].
// ---------------------------------------------------------------------------
DEVI v8f adj_inner_tile(const float* __restrict__ pcb, int i0, int j0) {
  const int lane = threadIdx.x & 31;
  const int h = lane >> 4;
  const int m = lane & 15;
  const float* rowi = pcb + (size_t)(i0 + m) * FIN_;
  const float* rowj = pcb + (size_t)(j0 + m) * FIN_;
  v8f acc = {0.f,0.f,0.f,0.f,0.f,0.f,0.f,0.f};
  // K chunk 0: coords c = 2h + {0,1}
  v2f a0 = { rowi[2*h + 0], rowi[2*h + 1] };
  v2f b0 = { rowj[2*h + 0], rowj[2*h + 1] };
  acc = __builtin_amdgcn_wmma_f32_16x16x4_f32(false, a0, false, b0,
                                              (short)0, acc, false, false);
  // K chunk 1: coords c = 4 + 2h + {0,1}; h==1 -> coords 6,7 (zero pad)
  v2f a1 = {0.f, 0.f};
  v2f b1 = {0.f, 0.f};
  if (h == 0) { a1[0] = rowi[4]; a1[1] = rowi[5];
                b1[0] = rowj[4]; b1[1] = rowj[5]; }
  acc = __builtin_amdgcn_wmma_f32_16x16x4_f32(false, a1, false, b1,
                                              (short)0, acc, false, false);
  return acc;
}

// Load one 16x16x32 bf16 A operand from a row-major row (contiguous 16B x2).
DEVI v16bf load_a32(const __bf16* __restrict__ rowp, int m0, int h) {
  v8bf lo = *(const v8bf*)(rowp + m0 + 8*h);
  v8bf hi = *(const v8bf*)(rowp + m0 + 8*h + 16);
  v16bf a;
#pragma unroll
  for (int e = 0; e < 8; ++e) { a[e] = lo[e]; a[e + 8] = hi[e]; }
  return a;
}

// Load one 16x16x32 bf16 B operand from a column stored contiguously.
DEVI v16bf load_b32(const __bf16* __restrict__ colp, int m0, int h) {
  v8bf lo = *(const v8bf*)(colp + m0 + 8*h);
  v8bf hi = *(const v8bf*)(colp + m0 + 8*h + 16);
  v16bf b;
#pragma unroll
  for (int e = 0; e < 8; ++e) { b[e] = lo[e]; b[e + 8] = hi[e]; }
  return b;
}

// ---------------------------------------------------------------------------
// init: sq[b,n] = ||x||^2 ; T0^T (bf16, [B][16][N], cols>=6 zero) ; Xt[:,0:6]=x
// ---------------------------------------------------------------------------
__global__ void k_init(const float* __restrict__ x, float* __restrict__ sq,
                       __bf16* __restrict__ T0t, float* __restrict__ Xt) {
  int idx = blockIdx.x * blockDim.x + threadIdx.x;   // B*N threads
  int b = idx / N_, n = idx % N_;
  const float* xr = x + (size_t)idx * FIN_;
  float v[FIN_]; float s = 0.f;
#pragma unroll
  for (int j = 0; j < FIN_; ++j) { v[j] = xr[j]; s += v[j] * v[j]; }
  sq[idx] = s;
#pragma unroll
  for (int j = 0; j < FIN_; ++j) Xt[(size_t)idx * KF_ + j] = v[j];
#pragma unroll
  for (int f = 0; f < 16; ++f)
    T0t[((size_t)b * 16 + f) * N_ + n] = (__bf16)(f < FIN_ ? v[f] : 0.f);
}

// ---------------------------------------------------------------------------
// degree: d[b,i] = sum_j exp(-dist(i,j)).  One block per (b, i-tile);
// 8 waves each cover 16 j-tiles. Deterministic shuffle+LDS reduction.
// ---------------------------------------------------------------------------
__global__ void k_degree(const float* __restrict__ x,
                         const float* __restrict__ sq,
                         float* __restrict__ deg) {
  int blk = blockIdx.x;
  int b = blk / NT_;
  int i0 = (blk % NT_) * 16;
  const float* pcb = x + (size_t)b * N_ * FIN_;
  const float* sqb = sq + (size_t)b * N_;
  int wave = threadIdx.x >> 5, lane = threadIdx.x & 31;
  int h = lane >> 4, m = lane & 15;

  float sqi[8], rs[8];
#pragma unroll
  for (int r = 0; r < 8; ++r) { sqi[r] = sqb[i0 + 8*h + r]; rs[r] = 0.f; }

  for (int t = 0; t < NT_ / 8; ++t) {          // 16 iterations per wave
    int j0 = (wave + 8 * t) * 16;
    v8f inner = adj_inner_tile(pcb, i0, j0);
    float sqj = sqb[j0 + m];
#pragma unroll
    for (int r = 0; r < 8; ++r)
      rs[r] += __expf(-(sqi[r] - 2.f * inner[r] + sqj));
  }
  // reduce across the 16 lanes of each half (fixed xor tree: deterministic)
#pragma unroll
  for (int r = 0; r < 8; ++r)
    for (int off = 8; off >= 1; off >>= 1) rs[r] += __shfl_xor(rs[r], off);

  __shared__ float lds[8][16];
  if (m == 0) {
#pragma unroll
    for (int r = 0; r < 8; ++r) lds[wave][8*h + r] = rs[r];
  }
  __syncthreads();
  if (threadIdx.x < 16) {
    float s = 0.f;
#pragma unroll
    for (int w = 0; w < 8; ++w) s += lds[w][threadIdx.x];
    deg[(size_t)b * N_ + i0 + threadIdx.x] = s;
  }
}

// ---------------------------------------------------------------------------
// buildL: L = I - dinv_i * exp(-dist) * dinv_j, stored bf16 row-major.
// L is symmetric, so a lane holding column j, rows i0+8h..+7 writes them as
// ROW j, cols i0+8h..+7 -> one contiguous 16B store per lane.
// ---------------------------------------------------------------------------
__global__ void k_buildL(const float* __restrict__ x,
                         const float* __restrict__ sq,
                         const float* __restrict__ deg,
                         __bf16* __restrict__ L) {
  int gw = blockIdx.x * (blockDim.x >> 5) + (threadIdx.x >> 5);
  int tj = gw % NT_;
  int ti = (gw / NT_) % NT_;
  int b  = gw / (NT_ * NT_);
  int i0 = ti * 16, j0 = tj * 16;
  const float* pcb = x   + (size_t)b * N_ * FIN_;
  const float* sqb = sq  + (size_t)b * N_;
  const float* db  = deg + (size_t)b * N_;
  int lane = threadIdx.x & 31, h = lane >> 4, m = lane & 15;

  v8f inner = adj_inner_tile(pcb, i0, j0);
  float sqj = sqb[j0 + m];
  float rdj = rsqrtf(db[j0 + m]);
  v8bf pack;
#pragma unroll
  for (int r = 0; r < 8; ++r) {
    int i = i0 + 8*h + r;
    float a  = __expf(-(sqb[i] - 2.f * inner[r] + sqj));
    float lv = -rsqrtf(db[i]) * a * rdj;
    if (i == j0 + m) lv += 1.0f;
    pack[r] = (__bf16)lv;
  }
  __bf16* Lrow = L + ((size_t)b * N_ + (j0 + m)) * N_;
  *(v8bf*)(Lrow + i0 + 8*h) = pack;   // 16B aligned contiguous store
}

// ---------------------------------------------------------------------------
// Lmul: OutT[b][f0+f][i] = alpha * (L @ Bt^T)[i][f] + beta * Xt[...p2col+f]
// bf16 WMMA 16x16x32, two independent accumulator chains (K chunks of 64),
// prefetch of the L row one chunk-stride ahead, deterministic LDS reduce.
// ---------------------------------------------------------------------------
__global__ void k_Lmul(const __bf16* __restrict__ L,
                       const __bf16* __restrict__ Bt, int brows,
                       float alpha, float beta,
                       const float* __restrict__ XtPrev, int p2col,
                       __bf16* __restrict__ OutT, int orows,
                       float* __restrict__ XtOut, int xtcol) {
  int blk = blockIdx.x;
  int b = blk / NT_;
  int i0 = (blk % NT_) * 16;
  int f0 = blockIdx.y * 16;
  int wave = threadIdx.x >> 5, lane = threadIdx.x & 31;
  int h = lane >> 4, m = lane & 15;

  const __bf16* La  = L  + ((size_t)b * N_ + (i0 + m)) * N_;
  const __bf16* Btb = Bt + ((size_t)b * brows + (f0 + m)) * N_;

  v8f acc0 = {0.f,0.f,0.f,0.f,0.f,0.f,0.f,0.f};
  v8f acc1 = {0.f,0.f,0.f,0.f,0.f,0.f,0.f,0.f};
  // K walked in chunks of 64 (two 32-deep WMMAs, independent chains)
  for (int t = wave; t < N_ / 64; t += 8) {
    int m0 = t * 64;
    __builtin_prefetch(La + m0 + 8 * 64, 0, 1);  // next chunk for this wave
    v16bf a0 = load_a32(La,  m0,      h);
    v16bf b0 = load_b32(Btb, m0,      h);
    acc0 = __builtin_amdgcn_wmma_f32_16x16x32_bf16(false, a0, false, b0,
                                                   (short)0, acc0, false, false);
    v16bf a1 = load_a32(La,  m0 + 32, h);
    v16bf b1 = load_b32(Btb, m0 + 32, h);
    acc1 = __builtin_amdgcn_wmma_f32_16x16x32_bf16(false, a1, false, b1,
                                                   (short)0, acc1, false, false);
  }

  __shared__ float red[8][256];
#pragma unroll
  for (int r = 0; r < 8; ++r)
    red[wave][(r + 8*h) * 16 + m] = acc0[r] + acc1[r];
  __syncthreads();

  int il = threadIdx.x >> 4;       // local row (i)
  int f  = threadIdx.x & 15;       // local col (feature)
  float s = 0.f;
#pragma unroll
  for (int w = 0; w < 8; ++w) s += red[w][threadIdx.x];
  s *= alpha;
  if (beta != 0.f && f < FIN_)
    s += beta * XtPrev[((size_t)b * N_ + i0 + il) * KF_ + p2col + f];
  OutT[((size_t)b * orows + f0 + f) * N_ + i0 + il] = (__bf16)s;
  if (xtcol >= 0 && f < FIN_)
    XtOut[((size_t)b * N_ + i0 + il) * KF_ + xtcol + f] = s;
}

// ---------------------------------------------------------------------------
// feat: out^T[b][f][n] = bf16(relu(Xt[b][n][:] @ W1[:,f] + b1[f]))
// ---------------------------------------------------------------------------
__global__ void k_feat(const float* __restrict__ Xt, const float* __restrict__ W1,
                       const float* __restrict__ b1, __bf16* __restrict__ OutT) {
  int idx = blockIdx.x * blockDim.x + threadIdx.x;   // B*N*FOUT
  int f = idx % FOUT_;
  int n = (idx / FOUT_) % N_;
  int b = idx / (FOUT_ * N_);
  const float* xr = Xt + ((size_t)b * N_ + n) * KF_;
  float s = b1[f];
#pragma unroll
  for (int j = 0; j < KF_; ++j) s += xr[j] * W1[j * FOUT_ + f];
  s = fmaxf(s, 0.f);
  OutT[((size_t)b * FOUT_ + f) * N_ + n] = (__bf16)s;
}

// ---------------------------------------------------------------------------
// quad: per 16x16 tile of (out^T @ Lout) accumulate sum of squares ->
// one deterministic partial per block.  K = n = 2048, dual accumulators.
// ---------------------------------------------------------------------------
__global__ void k_quad(const __bf16* __restrict__ OutT,
                       const __bf16* __restrict__ LoutT,
                       float* __restrict__ partial) {
  int blk = blockIdx.x;                 // B * 8 * 8
  int b = blk / 64;
  int t = blk % 64;
  int f0 = (t / 8) * 16, g0 = (t % 8) * 16;
  int wave = threadIdx.x >> 5, lane = threadIdx.x & 31;
  int h = lane >> 4, m = lane & 15;

  const __bf16* Ab = OutT  + ((size_t)b * FOUT_ + (f0 + m)) * N_;
  const __bf16* Bb = LoutT + ((size_t)b * FOUT_ + (g0 + m)) * N_;

  v8f acc0 = {0.f,0.f,0.f,0.f,0.f,0.f,0.f,0.f};
  v8f acc1 = {0.f,0.f,0.f,0.f,0.f,0.f,0.f,0.f};
  for (int tt = wave; tt < N_ / 64; tt += 8) {
    int m0 = tt * 64;
    __builtin_prefetch(Ab + m0 + 8 * 64, 0, 1);
    v16bf a0 = load_a32(Ab, m0,      h);
    v16bf b0 = load_b32(Bb, m0,      h);
    acc0 = __builtin_amdgcn_wmma_f32_16x16x32_bf16(false, a0, false, b0,
                                                   (short)0, acc0, false, false);
    v16bf a1 = load_a32(Ab, m0 + 32, h);
    v16bf b1 = load_b32(Bb, m0 + 32, h);
    acc1 = __builtin_amdgcn_wmma_f32_16x16x32_bf16(false, a1, false, b1,
                                                   (short)0, acc1, false, false);
  }
  float s = 0.f;
#pragma unroll
  for (int r = 0; r < 8; ++r) {
    float v = acc0[r] + acc1[r];
    s += v * v;
  }
  for (int off = 16; off >= 1; off >>= 1) s += __shfl_xor(s, off);
  __shared__ float wsum[8];
  if (lane == 0) wsum[wave] = s;
  __syncthreads();
  if (threadIdx.x == 0) {
    float tot = 0.f;
#pragma unroll
    for (int w = 0; w < 8; ++w) tot += wsum[w];
    partial[blockIdx.x] = tot;
  }
}

__global__ void k_reduce(const float* __restrict__ partial, int n,
                         float* __restrict__ out) {
  __shared__ float sh[256];
  float s = 0.f;
  for (int i = threadIdx.x; i < n; i += 256) s += partial[i];
  sh[threadIdx.x] = s;
  __syncthreads();
  for (int st = 128; st > 0; st >>= 1) {
    if ((int)threadIdx.x < st) sh[threadIdx.x] += sh[threadIdx.x + st];
    __syncthreads();
  }
  if (threadIdx.x == 0) out[0] = sh[0];
}

// ---------------------------------------------------------------------------
// head: pooled[b,f] = max_n out^T[b][f][n]; logits = pooled @ Wfc + bfc
// ---------------------------------------------------------------------------
__global__ void k_head(const __bf16* __restrict__ OutT,
                       const float* __restrict__ Wfc,
                       const float* __restrict__ bfc,
                       float* __restrict__ logits) {
  int b = blockIdx.x;
  int f = threadIdx.x;                    // 128 threads
  __shared__ float pooled[FOUT_];
  const v8bf* row = (const v8bf*)(OutT + ((size_t)b * FOUT_ + f) * N_);
  float mx = 0.f;                         // relu output is >= 0
  for (int c = 0; c < N_ / 8; ++c) {
    v8bf v = row[c];
#pragma unroll
    for (int e = 0; e < 8; ++e) mx = fmaxf(mx, (float)v[e]);
  }
  pooled[f] = mx;
  __syncthreads();
  if (f < CLS_) {
    float s = bfc[f];
#pragma unroll 8
    for (int j = 0; j < FOUT_; ++j) s += pooled[j] * Wfc[j * CLS_ + f];
    logits[b * CLS_ + f] = s;
  }
}

// ---------------------------------------------------------------------------
extern "C" void kernel_launch(void* const* d_in, const int* in_sizes, int n_in,
                              void* d_out, int out_size, void* d_ws, size_t ws_size,
                              hipStream_t stream) {
  const float* x   = (const float*)d_in[0];
  const float* W1  = (const float*)d_in[1];
  const float* b1  = (const float*)d_in[2];
  const float* Wfc = (const float*)d_in[3];
  const float* bfc = (const float*)d_in[4];
  float* out = (float*)d_out;

  char* p = (char*)d_ws;
  auto alloc = [&](size_t bytes) -> char* {
    char* r = p;
    p += (bytes + 255) & ~(size_t)255;
    return r;
  };
  float*  sq    = (float*) alloc(sizeof(float)  * (size_t)B_ * N_);
  float*  deg   = (float*) alloc(sizeof(float)  * (size_t)B_ * N_);
  __bf16* Ta    = (__bf16*)alloc(sizeof(__bf16) * (size_t)B_ * 16 * N_);
  __bf16* Tb    = (__bf16*)alloc(sizeof(__bf16) * (size_t)B_ * 16 * N_);
  float*  Xt    = (float*) alloc(sizeof(float)  * (size_t)B_ * N_ * KF_);
  __bf16* Lb    = (__bf16*)alloc(sizeof(__bf16) * (size_t)B_ * N_ * N_);
  __bf16* outT  = (__bf16*)alloc(sizeof(__bf16) * (size_t)B_ * FOUT_ * N_);
  __bf16* LoutT = (__bf16*)alloc(sizeof(__bf16) * (size_t)B_ * FOUT_ * N_);
  float*  part  = (float*) alloc(sizeof(float)  * (size_t)B_ * 64);

  k_init  <<<B_ * N_ / 256, 256, 0, stream>>>(x, sq, Ta, Xt);
  k_degree<<<B_ * NT_,      256, 0, stream>>>(x, sq, deg);
  k_buildL<<<B_ * NT_ * NT_ / 8, 256, 0, stream>>>(x, sq, deg, Lb);

  // Chebyshev recursion: T1 = L x ; Tk = 2 L T_{k-1} - T_{k-2}
  k_Lmul<<<dim3(B_ * NT_, 1), 256, 0, stream>>>(Lb, Ta, 16, 1.f,  0.f, Xt,  0, Tb, 16, Xt,  6);
  k_Lmul<<<dim3(B_ * NT_, 1), 256, 0, stream>>>(Lb, Tb, 16, 2.f, -1.f, Xt,  0, Ta, 16, Xt, 12);
  k_Lmul<<<dim3(B_ * NT_, 1), 256, 0, stream>>>(Lb, Ta, 16, 2.f, -1.f, Xt,  6, Tb, 16, Xt, 18);
  k_Lmul<<<dim3(B_ * NT_, 1), 256, 0, stream>>>(Lb, Tb, 16, 2.f, -1.f, Xt, 12, Ta, 16, Xt, 24);
  k_Lmul<<<dim3(B_ * NT_, 1), 256, 0, stream>>>(Lb, Ta, 16, 2.f, -1.f, Xt, 18, Tb, 16, Xt, 30);

  k_feat<<<B_ * N_ * FOUT_ / 256, 256, 0, stream>>>(Xt, W1, b1, outT);

  // Lout = L @ out  (8 feature tiles)
  k_Lmul<<<dim3(B_ * NT_, FOUT_ / 16), 256, 0, stream>>>(Lb, outT, FOUT_, 1.f, 0.f,
                                                         Xt, 0, LoutT, FOUT_, nullptr, -1);

  k_quad  <<<B_ * 64, 256, 0, stream>>>(outT, LoutT, part);
  k_reduce<<<1, 256, 0, stream>>>(part, B_ * 64, out + B_ * CLS_);
  k_head  <<<B_, 128, 0, stream>>>(outT, Wfc, bfc, out);
}
